// QuantizedLinear_13640816132862
// MI455X (gfx1250) — matrix operand order repaired and checked by reading, not compiled, not run on hardware
//
#include <hip/hip_runtime.h>

typedef __attribute__((ext_vector_type(16))) _Float16 v16h;
typedef __attribute__((ext_vector_type(8)))  _Float16 v8h;
typedef __attribute__((ext_vector_type(2)))  _Float16 v2h;
typedef __attribute__((ext_vector_type(8)))  float    v8f;

#define BM 256
#define BN 128
#define BK 128          // == GROUP_SIZE
#define ARS 136         // A row stride in f16 (BK + 8 pad); 272B = 17*16B
#define BRS 136         // B (per-column K vector) stride in f16
#define NTHREADS 512    // 16 wave32s: 8 (M) x 2 (N)

__global__ __launch_bounds__(NTHREADS)
void qlin_w4a16_wmma_kernel(const float* __restrict__ x,
                            const int*   __restrict__ qw,
                            const float* __restrict__ scales,
                            const float* __restrict__ zeros,
                            const float* __restrict__ bias,
                            float* __restrict__ out,
                            int M, int N, int K, int PK)
{
    __shared__ _Float16 lA[BM * ARS];   // x tile, f16, row-major (M x K-chunk)
    __shared__ _Float16 lB[BN * BRS];   // W tile, f16, per-output-col K vectors

    const int tid  = threadIdx.x;
    const int lane = tid & 31;
    const int wave = tid >> 5;
    const int l16  = lane & 15;
    const int lh   = lane >> 4;          // lane half: 0 or 1
    const int wm   = (wave & 7) * 32;    // wave M offset in block tile (0..224)
    const int wn   = (wave >> 3) * 64;   // wave N offset in block tile (0,64)

    const int n0 = blockIdx.x * BN;
    const int m0 = blockIdx.y * BM;

    v8f acc[2][4];
#pragma unroll
    for (int i = 0; i < 2; ++i)
#pragma unroll
        for (int j = 0; j < 4; ++j)
#pragma unroll
            for (int r = 0; r < 8; ++r) acc[i][j][r] = 0.0f;

    // x-tile loader: 512 thr cover 256x128 f32 -> 8 iters of (2 float4 -> 1 v8h)
    const int xr = tid >> 4;            // 0..31 (row base, step 32)
    const int xc = (tid & 15) * 8;      // 0..120 (col, 8 floats)

    // weight-tile loader: 4 threads per output column, 1 int4 (32 k) each
    const int bn  = tid >> 2;           // 0..127 (column within tile)
    const int bq  = tid & 3;            // which 32-k slice of the chunk
    const int bko = bq * 32;            // k offset (f16 elems) within chunk

    for (int k0 = 0; k0 < K; k0 += BK) {
        __syncthreads();   // previous chunk's compute done before overwrite

        // ---------- stage x tile: fp32 -> f16 into LDS ----------
#pragma unroll
        for (int r = 0; r < 8; ++r) {
            const int row = xr + r * 32;
            const float* xp = x + (long)(m0 + row) * K + k0 + xc;
            const float4 f0 = *(const float4*)(xp);
            const float4 f1 = *(const float4*)(xp + 4);
            v8h h;
            h[0] = (_Float16)f0.x; h[1] = (_Float16)f0.y;
            h[2] = (_Float16)f0.z; h[3] = (_Float16)f0.w;
            h[4] = (_Float16)f1.x; h[5] = (_Float16)f1.y;
            h[6] = (_Float16)f1.z; h[7] = (_Float16)f1.w;
            *(v8h*)(&lA[row * ARS + xc]) = h;   // one ds_store_b128
        }

        // ---------- stage weight tile: int4 -> f16 dequant into LDS ----------
        {
            const int g = k0 >> 7;                 // group (GROUP_SIZE==BK==128)
            const int n = n0 + bn;
            const float s = scales[(long)g * N + n];
            const float z = zeros [(long)g * N + n];
            v2h s2; s2.x = (_Float16)s;        s2.y = s2.x;
            v2h z2; z2.x = (_Float16)(-z * s); z2.y = z2.x;
            v2h kk; kk.x = (_Float16)(-1024.0f); kk.y = kk.x;

            const int4 q4 = *(const int4*)(qw + (long)n * PK + (k0 >> 3) + bq * 4);

            _Float16* dst = &lB[bn * BRS + bko];
            auto dq8 = [&](unsigned int q, int idx) {
                union { v2h h[4]; uint4 u; } o;
#pragma unroll
                for (int p = 0; p < 4; ++p) {
                    // pack two nibbles as f16 (1024+v0, 1024+v1)
                    unsigned int pk = (q & 0xFu) | ((q << 12) & 0x000F0000u) | 0x64006400u;
                    q >>= 8;
                    union { unsigned int u; v2h h; } cv; cv.u = pk;
                    v2h h = cv.h + kk;          // exact: (1024+v)-1024 = v
                    o.h[p] = h * s2 + z2;       // v_pk_fma_f16: v*s - z*s
                }
                *(uint4*)(dst + idx * 8) = o.u; // 8 f16 = one ds_store_b128
            };
            dq8((unsigned)q4.x, 0); dq8((unsigned)q4.y, 1);
            dq8((unsigned)q4.z, 2); dq8((unsigned)q4.w, 3);
        }
        __syncthreads();

        // ---------- compute: 4 WMMA K-steps of 32 over this chunk ----------
#pragma unroll
        for (int ks = 0; ks < 4; ++ks) {
            const int kb = ks * 32;
            v16h a[2], b[4];
#pragma unroll
            for (int i = 0; i < 2; ++i) {
                // A 16x32 f16 layout: lane<16 K{0..7,16..23}; lane>=16 K{8..15,24..31}
                const int row = wm + i * 16 + l16;
                const _Float16* ap = &lA[row * ARS + kb + lh * 8];
                union { v16h v; uint4 q[2]; } fa;
                fa.q[0] = *(const uint4*)(ap);
                fa.q[1] = *(const uint4*)(ap + 16);
                a[i] = fa.v;
            }
#pragma unroll
            for (int j = 0; j < 4; ++j) {
                // B 32x16 f16 layout: lane<16 K{0..7,16..23}; lane>=16 K{8..15,24..31}
                const int col = wn + j * 16 + l16;
                const _Float16* bp = &lB[col * BRS + kb + lh * 8];
                union { v16h v; uint4 q[2]; } fb;
                fb.q[0] = *(const uint4*)(bp);
                fb.q[1] = *(const uint4*)(bp + 16);
                b[j] = fb.v;
            }
#pragma unroll
            for (int i = 0; i < 2; ++i)
#pragma unroll
                for (int j = 0; j < 4; ++j)
                    acc[i][j] = __builtin_amdgcn_wmma_f32_16x16x32_f16(
                        false, a[i], false, b[j], (short)0, acc[i][j],
                        false, false);
        }
    }

    // ---------- epilogue: C/D layout (VGPR r -> M=r / r+8; N = lane%16) ----------
#pragma unroll
    for (int j = 0; j < 4; ++j) {
        const int n = n0 + wn + j * 16 + l16;
        const float bv = bias[n];
#pragma unroll
        for (int i = 0; i < 2; ++i) {
            const int mb = m0 + wm + i * 16 + lh * 8;
#pragma unroll
            for (int r = 0; r < 8; ++r) {
                out[(long)(mb + r) * N + n] = acc[i][j][r] + bv;
            }
        }
    }
}

extern "C" void kernel_launch(void* const* d_in, const int* in_sizes, int n_in,
                              void* d_out, int out_size, void* d_ws, size_t ws_size,
                              hipStream_t stream) {
    const float* x      = (const float*)d_in[0];
    const int*   qw     = (const int*)  d_in[1];
    const float* scales = (const float*)d_in[2];
    const float* zeros  = (const float*)d_in[3];
    const float* bias   = (const float*)d_in[4];
    float* out = (float*)d_out;

    const int N  = in_sizes[4];          // 11008
    const int PK = in_sizes[1] / N;      // 512 packed int32 per row
    const int K  = PK * 8;               // 4096 (8 int4 per int32)
    const int M  = in_sizes[0] / K;      // 8192 = 4*2048

    dim3 grid(N / BN, M / BM);           // 86 x 32 (exact)
    qlin_w4a16_wmma_kernel<<<grid, NTHREADS, 0, stream>>>(
        x, qw, scales, zeros, bias, out, M, N, K, PK);
}
